// SpatialEncoder3d_12111807775259
// MI455X (gfx1250) — matrix operand order repaired and checked by reading, not compiled, not run on hardware
//
#include <hip/hip_runtime.h>

typedef __attribute__((ext_vector_type(16))) _Float16 v16h;
typedef __attribute__((ext_vector_type(8)))  _Float16 v8h;
typedef __attribute__((ext_vector_type(8)))  float    v8f;
typedef __attribute__((ext_vector_type(4)))  float    v4f;

#define BATCH 4
#define NB    1024
#define KDIM  32
#define HDIM  8
#define MTYPE 100

#define LOG2E         1.44269504f
#define NEG_HALF_L2E  -0.72134752f      // -0.5 * log2(e)
#define LOG2_SQRT2PI  1.32574804f       // log2(sqrt(2*3.14159)); ref hard-codes pi=3.14159
#define GELU_2C       1.59576912f       // 2 * 0.7978845608
#define GELU_2C_L2E   -2.30212942f      // -GELU_2C * log2(e)

// tanh-form GELU as x * sigmoid(2c*(x + 0.044715 x^3)); branch-free,
// raw hw exp2/rcp (error ~1e-3, same order as the fp16 WMMA inputs).
__device__ __forceinline__ float gelu_fast(float v) {
    const float z = v * v;
    const float w = v * __builtin_fmaf(z, 0.044715f, 1.0f);
    const float e = __builtin_amdgcn_exp2f(GELU_2C_L2E * w);   // exp(-2c*w)
    return v * __builtin_amdgcn_rcpf(1.0f + e);                // v * sigmoid(2c*w)
}

__global__ __launch_bounds__(256, 1)
void spatial_encoder3d_kernel(
    const float* __restrict__ coord,      // (B, N, 3)
    const int*   __restrict__ node_type,  // (B, N)
    const float* __restrict__ means,      // (K)
    const float* __restrict__ stds,       // (K)
    const float* __restrict__ gamma_w,    // (2M+3, 1)
    const float* __restrict__ beta_w,     // (2M+3, 1)
    const float* __restrict__ W1,         // (K, K)
    const float* __restrict__ b1,         // (K)
    const float* __restrict__ W2,         // (H, K)
    const float* __restrict__ b2,         // (H)
    float*       __restrict__ out)        // (B, N, N, H)
{
    // per-wave LDS scratch: h relayout (16x32 f16 = 1KB) + out staging (16x8 f32 = 512B)
    __shared__ _Float16 h_lds[8][16][KDIM];
    __shared__ float    o_lds[8][16][HDIM];

    const int bi   = blockIdx.x;          // = b*NB + i
    const int b    = bi >> 10;
    const int lane = threadIdx.x & 31;
    const int wave = threadIdx.x >> 5;    // 0..7
    const int half = lane >> 4;           // 0 or 1
    const int l16  = lane & 15;

    // ---- per-lane Gaussian params in WMMA f16 A-layout order -------------
    // A-layout (16x32 f16): lane L row M=L%16; element e -> k = 8*half + e + (e>=8 ? 8 : 0)
    // gk = exp2(qq*u^2 + lc), u = x - mean;  qq = -0.5*log2e/s^2, lc = -log2(sqrt(2pi)) - log2(s)
    float mean_l[16], qq_l[16], lc_l[16];
    #pragma unroll
    for (int e = 0; e < 16; ++e) {
        const int   k  = half * 8 + e + ((e >= 8) ? 8 : 0);
        const float s  = fabsf(stds[k]) + 0.01f;
        const float is = __builtin_amdgcn_rcpf(s);
        mean_l[e] = means[k];
        qq_l[e]   = NEG_HALF_L2E * is * is;
        lc_l[e]   = -LOG2_SQRT2PI - __builtin_amdgcn_logf(s);  // v_log_f32 is log2
    }

    // ---- weight fragments in WMMA f16 B-layout --------------------------
    // B-layout (32x16 f16): lane L column n=L%16; element e -> k = 8*half + e + (e>=8 ? 8 : 0)
    v16h bw1a, bw1b, bw2;
    const int klo = half * 16;
    #pragma unroll
    for (int e = 0; e < 16; ++e) {
        const int k = half * 8 + e + ((e >= 8) ? 8 : 0);
        bw1a[e] = (_Float16)W1[l16 * KDIM + k];          // h cols 0..15
        bw1b[e] = (_Float16)W1[(16 + l16) * KDIM + k];   // h cols 16..31
        bw2[e]  = (l16 < HDIM) ? (_Float16)W2[l16 * KDIM + k] : (_Float16)0.0f;
    }

    // ---- per-row (b,i) scalars ------------------------------------------
    const float cx = coord[bi * 3 + 0];
    const float cy = coord[bi * 3 + 1];
    const float cz = coord[bi * 3 + 2];
    const int   nt_i   = node_type[bi];
    const float g_src  = gamma_w[nt_i + 2];
    const float be_src = beta_w [nt_i + 2];
    const float bias0  = b1[l16];
    const float bias1  = b1[16 + l16];
    const float bias2  = (l16 < HDIM) ? b2[l16] : 0.0f;

    // ---- 8 tiles of 16 consecutive j per wave ---------------------------
    for (int t = 0; t < 8; ++t) {
        const int j0 = wave * 128 + t * 16;
        const int j  = j0 + l16;

        // distance + affine (type-pair gather)
        const float dx = cx - coord[(b * NB + j) * 3 + 0];
        const float dy = cy - coord[(b * NB + j) * 3 + 1];
        const float dz = cz - coord[(b * NB + j) * 3 + 2];
        const float dist = __builtin_amdgcn_sqrtf(dx * dx + dy * dy + dz * dz);
        const int   nt_j = node_type[b * NB + j];
        const float gmm  = g_src  + gamma_w[nt_j + MTYPE + 3];
        const float bet  = be_src + beta_w [nt_j + MTYPE + 3];
        const float x    = gmm * dist + bet;

        // Gaussian basis, built directly in f16 A-layout: one fused exp2 per element
        v16h a;
        #pragma unroll
        for (int e = 0; e < 16; ++e) {
            const float u  = x - mean_l[e];
            const float u2 = u * u;
            a[e] = (_Float16)__builtin_amdgcn_exp2f(__builtin_fmaf(u2, qq_l[e], lc_l[e]));
        }

        // GEMM1: h = gk @ W1^T  (two 16-col halves)
        v8f c0 = {}, c1 = {};
        c0 = __builtin_amdgcn_wmma_f32_16x16x32_f16(false, a, false, bw1a, (short)0, c0, false, false);
        c1 = __builtin_amdgcn_wmma_f32_16x16x32_f16(false, a, false, bw1b, (short)0, c1, false, false);

        // bias + GELU, stage h into LDS (C-layout: lane holds col n=l16, rows m=8*half+r)
        #pragma unroll
        for (int r = 0; r < 8; ++r) {
            const int m = half * 8 + r;
            h_lds[wave][m][l16]      = (_Float16)gelu_fast(c0[r] + bias0);
            h_lds[wave][m][16 + l16] = (_Float16)gelu_fast(c1[r] + bias1);
        }
        __builtin_amdgcn_wave_barrier();   // same-wave DS ops are in-order; stop compiler reordering

        // reload h in A-layout: row M=l16, elements 0..7 -> k=8*half+e, 8..15 -> k=8*half+16+e
        const _Float16* rowp = &h_lds[wave][l16][0];
        const v8h lo = *(const v8h*)(rowp + half * 8);
        const v8h hi = *(const v8h*)(rowp + half * 8 + 16);
        v16h a2;
        #pragma unroll
        for (int e = 0; e < 8; ++e) { a2[e] = lo[e]; a2[8 + e] = hi[e]; }

        // GEMM2: out = h @ W2^T (B cols 8..15 are zero padding)
        v8f d = {};
        d = __builtin_amdgcn_wmma_f32_16x16x32_f16(false, a2, false, bw2, (short)0, d, false, false);

        // stage valid 16x8 block, then one coalesced b128 store per lane
        if (l16 < HDIM) {
            #pragma unroll
            for (int r = 0; r < 8; ++r)
                o_lds[wave][half * 8 + r][l16] = d[r] + bias2;
        }
        __builtin_amdgcn_wave_barrier();

        const float* op = &o_lds[wave][0][0];
        const v4f val = *(const v4f*)(op + lane * 4);
        *(v4f*)(out + ((size_t)bi * NB + j0) * HDIM + lane * 4) = val;
        __builtin_amdgcn_wave_barrier();   // WAR guard before next tile reuses LDS
    }
}

extern "C" void kernel_launch(void* const* d_in, const int* in_sizes, int n_in,
                              void* d_out, int out_size, void* d_ws, size_t ws_size,
                              hipStream_t stream) {
    const float* coord     = (const float*)d_in[0];
    const int*   node_type = (const int*)  d_in[1];
    const float* means     = (const float*)d_in[2];
    const float* stds      = (const float*)d_in[3];
    const float* gamma_w   = (const float*)d_in[4];
    const float* beta_w    = (const float*)d_in[5];
    const float* W1        = (const float*)d_in[6];
    const float* b1        = (const float*)d_in[7];
    const float* W2        = (const float*)d_in[8];
    const float* b2        = (const float*)d_in[9];
    float*       out       = (float*)d_out;

    dim3 grid(BATCH * NB);   // one block per (b, i) row
    dim3 block(256);         // 8 waves; each wave covers 128 j's (8 tiles of 16)
    spatial_encoder3d_kernel<<<grid, block, 0, stream>>>(
        coord, node_type, means, stds, gamma_w, beta_w, W1, b1, W2, b2, out);
}
